// HGTEncoder_88605175316549
// MI455X (gfx1250) — matrix operand order repaired and checked by reading, not compiled, not run on hardware
//
#include <hip/hip_runtime.h>
#include <hip/hip_bf16.h>
#include <math.h>

// ---------------- problem constants (match reference) ----------------
#define NGENE 40000
#define NPEAK 80000
#define NEDGE 150000
#define HID 256
#define NH 4
#define HD 64
#define NLAYER 2
#define INPG 128
#define INPP 64
#define SCALE_ATT 0.125f   // 1/sqrt(64)

typedef __bf16 bf16;
typedef __attribute__((ext_vector_type(16))) __bf16 v16bf;
typedef __attribute__((ext_vector_type(8)))  __bf16 v8bf;
typedef __attribute__((ext_vector_type(8)))  float  v8f;

// ---------------- helpers ----------------
__device__ __forceinline__ float gelu_tanh(float x) {
    const float c0 = 0.7978845608028654f; // sqrt(2/pi)
    return 0.5f * x * (1.0f + tanhf(c0 * (x + 0.044715f * x * x * x)));
}

__device__ __forceinline__ void atomicMaxF32(float* addr, float val) {
    if (val >= 0.0f) atomicMax((int*)addr, __float_as_int(val));
    else             atomicMin((unsigned int*)addr, (unsigned int)__float_as_int(val));
}

// Packed B layout: element (k, n) of a [K x N] weight lives at
//   ((kt*(N/16) + nt)*32 + lane)*16 + i
// with kt=k/32, kl=k%32, lane=(n&15) + (kl>=16 ? 16 : 0), i=kl&15, nt=n/16.
// This is a contiguous-k B order (k = 16*(lane/16) + i), matching the A reads, so a wave
// fetches one 16x16 fragment as a contiguous 32B load per lane.
__device__ __forceinline__ size_t packedB_idx(int k, int n, int N) {
    int kt = k >> 5;
    int kl = k & 31;
    int lane = (n & 15) + ((kl & 16) ? 16 : 0);
    int i = kl & 15;
    int nt = n >> 4;
    return (((size_t)kt * (N >> 4) + nt) * 32 + (size_t)lane) * 16 + (size_t)i;
}

// ---------------- generic bf16 WMMA GEMM ----------------
// C[M,N] = epilogue( pre(A)[M,K] @ B[K,N] + bias[N] ),  B given in packed layout.
// PRE_GELU: apply tanh-gelu to A elements while staging to LDS.
// EPI: 0 = bias only, 1 = bias+relu, 2 = relu(sig(skip)*(acc+bias) + (1-sig)*resid)
#define TM 128
#define TN 64
#define TK 32

template <int PRE_GELU, int EPI>
__global__ __launch_bounds__(256) void gemm_wmma(
    const float* __restrict__ A, const bf16* __restrict__ Bp,
    const float* __restrict__ bias, const float* __restrict__ resid,
    const float* __restrict__ skipLogit, float* __restrict__ C,
    int M, int N, int K)
{
    __shared__ bf16 As[2][TM][TK];   // 16 KB, double-buffered

    const int tid  = threadIdx.x;
    const int lane = tid & 31;
    const int wave = tid >> 5;                 // 0..7 -> 16-row strip
    const int m0 = blockIdx.x * TM;
    const int n0 = blockIdx.y * TN;
    const int ntiles = N >> 4;

    v8f acc[4] = {};

    // staging coordinates: each thread handles 16 contiguous columns of one row
    const int sr = tid >> 1;
    const int sc = (tid & 1) * 16;

    auto stageA = [&](int buf, int k0) {
        float vals[16];
        int gr = m0 + sr;
        if (gr < M) {
            const float4* ap4 = (const float4*)(A + (size_t)gr * K + k0 + sc);
            float4 f0 = ap4[0], f1 = ap4[1], f2 = ap4[2], f3 = ap4[3];
            vals[0]=f0.x; vals[1]=f0.y; vals[2]=f0.z; vals[3]=f0.w;
            vals[4]=f1.x; vals[5]=f1.y; vals[6]=f1.z; vals[7]=f1.w;
            vals[8]=f2.x; vals[9]=f2.y; vals[10]=f2.z; vals[11]=f2.w;
            vals[12]=f3.x; vals[13]=f3.y; vals[14]=f3.z; vals[15]=f3.w;
        } else {
            #pragma unroll
            for (int i = 0; i < 16; ++i) vals[i] = 0.0f;
        }
        if (PRE_GELU) {
            #pragma unroll
            for (int i = 0; i < 16; ++i) vals[i] = gelu_tanh(vals[i]);
        }
        v8bf lo, hi;
        #pragma unroll
        for (int i = 0; i < 8; ++i) { lo[i] = (bf16)vals[i]; hi[i] = (bf16)vals[8 + i]; }
        *(v8bf*)&As[buf][sr][sc]     = lo;
        *(v8bf*)&As[buf][sr][sc + 8] = hi;
    };

    const int am   = lane & 15;
    const int arow = wave * 16 + am;
    const int koff = (lane < 16) ? 0 : 16;
    const v16bf* bb = (const v16bf*)Bp;
    const int nk = K / TK;

    stageA(0, 0);
    __syncthreads();

    for (int kt = 0; kt < nk; ++kt) {
        const int buf = kt & 1;

        // A fragment: two aligned 16B LDS loads, concat to v16bf
        v8bf alo = *(const v8bf*)&As[buf][arow][koff];
        v8bf ahi = *(const v8bf*)&As[buf][arow][koff + 8];
        v16bf af = __builtin_shufflevector(alo, ahi,
            0,1,2,3,4,5,6,7,8,9,10,11,12,13,14,15);

        // B fragments: hoist all 4 loads (32B contiguous per lane each) so
        // the waits can be partial and loads overlap the WMMAs below
        v16bf bfr[4];
        #pragma unroll
        for (int t = 0; t < 4; ++t)
            bfr[t] = bb[((size_t)kt * ntiles + (n0 >> 4) + t) * 32 + lane];

        // stage next A tile into the other buffer while loads are in flight
        if (kt + 1 < nk) stageA(buf ^ 1, (kt + 1) * TK);

        #pragma unroll
        for (int t = 0; t < 4; ++t)
            acc[t] = __builtin_amdgcn_wmma_f32_16x16x32_bf16(
                false, af, false, bfr[t], (short)0, acc[t], false, false);

        __syncthreads();
    }

    // ---- epilogue ----
    float a = 1.0f, b = 0.0f;
    if (EPI == 2) {
        float s = skipLogit[0];
        a = 1.0f / (1.0f + __expf(-s));
        b = 1.0f - a;
    }
    const int nl = lane & 15;
    const int mb = (lane < 16) ? 0 : 8;
    #pragma unroll
    for (int t = 0; t < 4; ++t) {
        int gn = n0 + t * 16 + nl;
        float bi = bias[gn];
        #pragma unroll
        for (int v = 0; v < 8; ++v) {
            int gm = m0 + wave * 16 + mb + v;
            if (gm < M) {
                float val = acc[t][v] + bi;
                if (EPI == 1) val = fmaxf(val, 0.0f);
                if (EPI == 2) {
                    val = a * val + b * resid[(size_t)gm * N + gn];
                    val = fmaxf(val, 0.0f);
                }
                C[(size_t)gm * N + gn] = val;
            }
        }
    }
}

// ---------------- small utility kernels ----------------
__global__ __launch_bounds__(256) void fill_kernel(float* p, float v, int n) {
    int i = blockIdx.x * 256 + threadIdx.x;
    if (i < n) p[i] = v;
}

// repack fp32 row-major [K x N] weight into packed bf16 B-fragment layout
__global__ __launch_bounds__(256) void repack_w_kernel(
    const float* __restrict__ W, bf16* __restrict__ out, int K, int N)
{
    int idx = blockIdx.x * 256 + threadIdx.x;
    if (idx >= K * N) return;
    int k = idx / N, n = idx - k * N;
    out[packedB_idx(k, n, N)] = (bf16)W[idx];
}

// Wout[i, h*64+f] = sum_d W[i, h*64+d] * rel[h, d, f]  -> packed bf16 directly
__global__ __launch_bounds__(256) void combine_w_kernel(
    const float* __restrict__ W, const float* __restrict__ rel, bf16* __restrict__ out)
{
    int idx = blockIdx.x * 256 + threadIdx.x;
    if (idx >= HID * HID) return;
    int i = idx >> 8, col = idx & 255;
    int h = col >> 6, f = col & 63;
    const float* wr = W + (size_t)i * HID + h * HD;
    const float* rr = rel + (size_t)h * HD * HD + f;
    float s = 0.0f;
    #pragma unroll 4
    for (int d = 0; d < HD; ++d) s += wr[d] * rr[(size_t)d * HD];
    out[packedB_idx(i, col, HID)] = (bf16)s;
}

// bout[h*64+f] = sum_d b[h*64+d] * rel[h, d, f]
__global__ __launch_bounds__(256) void combine_b_kernel(
    const float* __restrict__ b, const float* __restrict__ rel, float* __restrict__ out)
{
    int col = threadIdx.x;
    int h = col >> 6, f = col & 63;
    const float* br = b + h * HD;
    const float* rr = rel + (size_t)h * HD * HD + f;
    float s = 0.0f;
    for (int d = 0; d < HD; ++d) s += br[d] * rr[(size_t)d * HD];
    out[col] = s;
}

// ---------------- edge kernels (one wave32 per edge) ----------------
__global__ __launch_bounds__(256) void edge_logits_kernel(
    const int* __restrict__ ei, const float* __restrict__ kt, const float* __restrict__ q,
    const float* __restrict__ prel, float* __restrict__ lg, float* __restrict__ mx, int E)
{
    int e = blockIdx.x * 8 + (threadIdx.x >> 5);
    if (e >= E) return;
    int lane = threadIdx.x & 31;
    int src = ei[e], dst = ei[E + e];
    const float4* kp = (const float4*)(kt + (size_t)src * HID + lane * 8);
    const float4* qp = (const float4*)(q  + (size_t)dst * HID + lane * 8);
    float4 k0 = kp[0], k1 = kp[1], q0 = qp[0], q1 = qp[1];
    float s = k0.x*q0.x + k0.y*q0.y + k0.z*q0.z + k0.w*q0.w
            + k1.x*q1.x + k1.y*q1.y + k1.z*q1.z + k1.w*q1.w;
    s += __shfl_down(s, 4, 32);
    s += __shfl_down(s, 2, 32);
    s += __shfl_down(s, 1, 32);
    if ((lane & 7) == 0) {
        int h = lane >> 3;
        float v = s * prel[h] * SCALE_ATT;
        lg[(size_t)e * NH + h] = v;
        atomicMaxF32(&mx[(size_t)dst * NH + h], v);
    }
}

__global__ __launch_bounds__(256) void edge_den_kernel(
    const int* __restrict__ ei, const float* __restrict__ mx,
    const float* __restrict__ lg, float* __restrict__ den, int E)
{
    int idx = blockIdx.x * 256 + threadIdx.x;
    if (idx >= E * NH) return;
    int e = idx >> 2, h = idx & 3;
    int dst = ei[E + e];
    atomicAdd(&den[(size_t)dst * NH + h], __expf(lg[idx] - mx[(size_t)dst * NH + h]));
}

__global__ __launch_bounds__(256) void edge_agg_kernel(
    const int* __restrict__ ei, const float* __restrict__ lg,
    const float* __restrict__ mx, const float* __restrict__ den,
    const float* __restrict__ vt, float* __restrict__ agg, int E)
{
    int e = blockIdx.x * 8 + (threadIdx.x >> 5);
    if (e >= E) return;
    int lane = threadIdx.x & 31;
    int h = lane >> 3;
    int src = ei[e], dst = ei[E + e];
    float a = __expf(lg[(size_t)e * NH + h] - mx[(size_t)dst * NH + h])
              / den[(size_t)dst * NH + h];
    const float4* vp = (const float4*)(vt + (size_t)src * HID + lane * 8);
    float*        ap = agg + (size_t)dst * HID + lane * 8;
    float4 v0 = vp[0], v1 = vp[1];
    atomicAdd(&ap[0], a * v0.x); atomicAdd(&ap[1], a * v0.y);
    atomicAdd(&ap[2], a * v0.z); atomicAdd(&ap[3], a * v0.w);
    atomicAdd(&ap[4], a * v1.x); atomicAdd(&ap[5], a * v1.y);
    atomicAdd(&ap[6], a * v1.z); atomicAdd(&ap[7], a * v1.w);
}

// ---------------- host orchestration ----------------
static inline size_t al256(size_t x) { return (x + 255) & ~(size_t)255; }

extern "C" void kernel_launch(void* const* d_in, const int* in_sizes, int n_in,
                              void* d_out, int out_size, void* d_ws, size_t ws_size,
                              hipStream_t stream) {
    (void)in_sizes; (void)n_in; (void)out_size; (void)ws_size;
    const float* x_gene  = (const float*)d_in[0];
    const float* x_peak  = (const float*)d_in[1];
    const int*   ei_gp   = (const int*)d_in[2];
    const int*   ei_pg   = (const int*)d_in[3];
    const int*   ei_pp   = (const int*)d_in[4];
    const float* W_in_g  = (const float*)d_in[5];
    const float* b_in_g  = (const float*)d_in[6];
    const float* W_in_p  = (const float*)d_in[7];
    const float* b_in_p  = (const float*)d_in[8];
    const float* Wk      = (const float*)d_in[9];
    const float* bk      = (const float*)d_in[10];
    const float* Wq      = (const float*)d_in[11];
    const float* bq      = (const float*)d_in[12];
    const float* Wv      = (const float*)d_in[13];
    const float* bv      = (const float*)d_in[14];
    const float* a_rel   = (const float*)d_in[15];
    const float* m_rel   = (const float*)d_in[16];
    const float* p_rel   = (const float*)d_in[17];
    const float* W_oc    = (const float*)d_in[18];
    const float* b_oc    = (const float*)d_in[19];
    const float* skip    = (const float*)d_in[20];
    const float* W_out_g = (const float*)d_in[21];
    const float* b_out_g = (const float*)d_in[22];
    const float* W_out_p = (const float*)d_in[23];
    const float* b_out_p = (const float*)d_in[24];
    float* out = (float*)d_out;

    char* ws = (char*)d_ws;
    size_t off = 0;
    auto carve = [&](size_t bytes) -> void* { void* p = ws + off; off += al256(bytes); return p; };

    const size_t WSZ = (size_t)HID * HID;
    bf16* Wing_bf  = (bf16*)carve(INPG * HID * sizeof(bf16));
    bf16* Winp_bf  = (bf16*)carve(INPP * HID * sizeof(bf16));
    bf16* Wq_bf    = (bf16*)carve(NLAYER * 2 * WSZ * sizeof(bf16));
    bf16* Woc_bf   = (bf16*)carve(NLAYER * 2 * WSZ * sizeof(bf16));
    bf16* Woutg_bf = (bf16*)carve(WSZ * sizeof(bf16));
    bf16* Woutp_bf = (bf16*)carve(WSZ * sizeof(bf16));
    bf16* Wka_bf   = (bf16*)carve(NLAYER * 3 * WSZ * sizeof(bf16));
    bf16* Wvm_bf   = (bf16*)carve(NLAYER * 3 * WSZ * sizeof(bf16));
    float* bka     = (float*)carve(NLAYER * 3 * HID * sizeof(float));
    float* bvm     = (float*)carve(NLAYER * 3 * HID * sizeof(float));

    float* xg[2] = { (float*)carve((size_t)NGENE * HID * 4), (float*)carve((size_t)NGENE * HID * 4) };
    float* xp[2] = { (float*)carve((size_t)NPEAK * HID * 4), (float*)carve((size_t)NPEAK * HID * 4) };
    float* qg   = (float*)carve((size_t)NGENE * HID * 4);
    float* qp   = (float*)carve((size_t)NPEAK * HID * 4);
    float* ktg  = (float*)carve((size_t)NGENE * HID * 4);
    float* vtg  = (float*)carve((size_t)NGENE * HID * 4);
    float* ktp1 = (float*)carve((size_t)NPEAK * HID * 4);
    float* ktp2 = (float*)carve((size_t)NPEAK * HID * 4);
    float* vtp1 = (float*)carve((size_t)NPEAK * HID * 4);
    float* vtp2 = (float*)carve((size_t)NPEAK * HID * 4);
    float* aggg = (float*)carve((size_t)NGENE * HID * 4);
    float* aggp = (float*)carve((size_t)NPEAK * HID * 4);
    float* mxg  = (float*)carve((size_t)NGENE * NH * 4);
    float* deng = (float*)carve((size_t)NGENE * NH * 4);
    float* mxp  = (float*)carve((size_t)NPEAK * NH * 4);
    float* denp = (float*)carve((size_t)NPEAK * NH * 4);
    float* lg0  = (float*)carve((size_t)NEDGE * NH * 4);
    float* lg1  = (float*)carve((size_t)NEDGE * NH * 4);
    float* lg2  = (float*)carve((size_t)NEDGE * NH * 4);

    dim3 blk(256);
    auto cdiv = [](int a, int b) { return (a + b - 1) / b; };
    auto ggrid = [&](int M) { return dim3(cdiv(M, TM), HID / TN); };

    // ---- weight repack (fp32 -> packed bf16 B-fragment layout) ----
    hipLaunchKernelGGL(repack_w_kernel, dim3(cdiv(INPG*HID,256)), blk, 0, stream, W_in_g, Wing_bf, INPG, HID);
    hipLaunchKernelGGL(repack_w_kernel, dim3(cdiv(INPP*HID,256)), blk, 0, stream, W_in_p, Winp_bf, INPP, HID);
    for (int i = 0; i < NLAYER * 2; ++i) {
        hipLaunchKernelGGL(repack_w_kernel, dim3(cdiv((int)WSZ,256)), blk, 0, stream,
            Wq + (size_t)i * WSZ, Wq_bf + (size_t)i * WSZ, HID, HID);
        hipLaunchKernelGGL(repack_w_kernel, dim3(cdiv((int)WSZ,256)), blk, 0, stream,
            W_oc + (size_t)i * WSZ, Woc_bf + (size_t)i * WSZ, HID, HID);
    }
    hipLaunchKernelGGL(repack_w_kernel, dim3(cdiv((int)WSZ,256)), blk, 0, stream, W_out_g, Woutg_bf, HID, HID);
    hipLaunchKernelGGL(repack_w_kernel, dim3(cdiv((int)WSZ,256)), blk, 0, stream, W_out_p, Woutp_bf, HID, HID);

    // ---- relation-fused weight precombination (writes packed bf16) ----
    for (int l = 0; l < NLAYER; ++l) {
        for (int r = 0; r < 3; ++r) {
            int st = (r == 0) ? 0 : 1;
            const float* relA = a_rel + (size_t)(l * 3 + r) * NH * HD * HD;
            const float* relM = m_rel + (size_t)(l * 3 + r) * NH * HD * HD;
            hipLaunchKernelGGL(combine_w_kernel, dim3(cdiv(HID*HID,256)), blk, 0, stream,
                Wk + (size_t)(l*2+st)*WSZ, relA, Wka_bf + (size_t)(l*3+r)*WSZ);
            hipLaunchKernelGGL(combine_b_kernel, dim3(1), blk, 0, stream,
                bk + (size_t)(l*2+st)*HID, relA, bka + (size_t)(l*3+r)*HID);
            hipLaunchKernelGGL(combine_w_kernel, dim3(cdiv(HID*HID,256)), blk, 0, stream,
                Wv + (size_t)(l*2+st)*WSZ, relM, Wvm_bf + (size_t)(l*3+r)*WSZ);
            hipLaunchKernelGGL(combine_b_kernel, dim3(1), blk, 0, stream,
                bv + (size_t)(l*2+st)*HID, relM, bvm + (size_t)(l*3+r)*HID);
        }
    }

    // ---- input projections (relu) ----
    hipLaunchKernelGGL((gemm_wmma<0,1>), ggrid(NGENE), blk, 0, stream,
        x_gene, Wing_bf, b_in_g, nullptr, nullptr, xg[0], NGENE, HID, INPG);
    hipLaunchKernelGGL((gemm_wmma<0,1>), ggrid(NPEAK), blk, 0, stream,
        x_peak, Winp_bf, b_in_p, nullptr, nullptr, xp[0], NPEAK, HID, INPP);

    int cur = 0;
    for (int l = 0; l < NLAYER; ++l) {
        const float* xgc = xg[cur];
        const float* xpc = xp[cur];
        float* xgn = xg[1 - cur];
        float* xpn = xp[1 - cur];

        // Q / relation-fused K_t / V_t projections
        hipLaunchKernelGGL((gemm_wmma<0,0>), ggrid(NGENE), blk, 0, stream,
            xgc, Wq_bf + (size_t)(l*2+0)*WSZ, bq + (size_t)(l*2+0)*HID, nullptr, nullptr, qg, NGENE, HID, HID);
        hipLaunchKernelGGL((gemm_wmma<0,0>), ggrid(NPEAK), blk, 0, stream,
            xpc, Wq_bf + (size_t)(l*2+1)*WSZ, bq + (size_t)(l*2+1)*HID, nullptr, nullptr, qp, NPEAK, HID, HID);
        hipLaunchKernelGGL((gemm_wmma<0,0>), ggrid(NGENE), blk, 0, stream,
            xgc, Wka_bf + (size_t)(l*3+0)*WSZ, bka + (size_t)(l*3+0)*HID, nullptr, nullptr, ktg, NGENE, HID, HID);
        hipLaunchKernelGGL((gemm_wmma<0,0>), ggrid(NGENE), blk, 0, stream,
            xgc, Wvm_bf + (size_t)(l*3+0)*WSZ, bvm + (size_t)(l*3+0)*HID, nullptr, nullptr, vtg, NGENE, HID, HID);
        hipLaunchKernelGGL((gemm_wmma<0,0>), ggrid(NPEAK), blk, 0, stream,
            xpc, Wka_bf + (size_t)(l*3+1)*WSZ, bka + (size_t)(l*3+1)*HID, nullptr, nullptr, ktp1, NPEAK, HID, HID);
        hipLaunchKernelGGL((gemm_wmma<0,0>), ggrid(NPEAK), blk, 0, stream,
            xpc, Wvm_bf + (size_t)(l*3+1)*WSZ, bvm + (size_t)(l*3+1)*HID, nullptr, nullptr, vtp1, NPEAK, HID, HID);
        hipLaunchKernelGGL((gemm_wmma<0,0>), ggrid(NPEAK), blk, 0, stream,
            xpc, Wka_bf + (size_t)(l*3+2)*WSZ, bka + (size_t)(l*3+2)*HID, nullptr, nullptr, ktp2, NPEAK, HID, HID);
        hipLaunchKernelGGL((gemm_wmma<0,0>), ggrid(NPEAK), blk, 0, stream,
            xpc, Wvm_bf + (size_t)(l*3+2)*WSZ, bvm + (size_t)(l*3+2)*HID, nullptr, nullptr, vtp2, NPEAK, HID, HID);

        // reset softmax / aggregation state (every call -> replay-safe)
        hipLaunchKernelGGL(fill_kernel, dim3(cdiv(NGENE*NH,256)), blk, 0, stream, mxg, -INFINITY, NGENE*NH);
        hipLaunchKernelGGL(fill_kernel, dim3(cdiv(NGENE*NH,256)), blk, 0, stream, deng, 0.0f, NGENE*NH);
        hipLaunchKernelGGL(fill_kernel, dim3(cdiv(NPEAK*NH,256)), blk, 0, stream, mxp, -INFINITY, NPEAK*NH);
        hipLaunchKernelGGL(fill_kernel, dim3(cdiv(NPEAK*NH,256)), blk, 0, stream, denp, 0.0f, NPEAK*NH);
        hipLaunchKernelGGL(fill_kernel, dim3(cdiv(NGENE*HID,256)), blk, 0, stream, aggg, 0.0f, NGENE*HID);
        hipLaunchKernelGGL(fill_kernel, dim3(cdiv(NPEAK*HID,256)), blk, 0, stream, aggp, 0.0f, NPEAK*HID);

        dim3 egrid(cdiv(NEDGE, 8));
        dim3 dgrid(cdiv(NEDGE * NH, 256));
        const float* pr0 = p_rel + (size_t)(l*3+0)*NH;
        const float* pr1 = p_rel + (size_t)(l*3+1)*NH;
        const float* pr2 = p_rel + (size_t)(l*3+2)*NH;

        // logits + running max  (r0: gene->peak, r1: peak->gene, r2: peak->peak)
        hipLaunchKernelGGL(edge_logits_kernel, egrid, blk, 0, stream, ei_gp, ktg,  qp, pr0, lg0, mxp, NEDGE);
        hipLaunchKernelGGL(edge_logits_kernel, egrid, blk, 0, stream, ei_pg, ktp1, qg, pr1, lg1, mxg, NEDGE);
        hipLaunchKernelGGL(edge_logits_kernel, egrid, blk, 0, stream, ei_pp, ktp2, qp, pr2, lg2, mxp, NEDGE);
        // denominators
        hipLaunchKernelGGL(edge_den_kernel, dgrid, blk, 0, stream, ei_gp, mxp, lg0, denp, NEDGE);
        hipLaunchKernelGGL(edge_den_kernel, dgrid, blk, 0, stream, ei_pg, mxg, lg1, deng, NEDGE);
        hipLaunchKernelGGL(edge_den_kernel, dgrid, blk, 0, stream, ei_pp, mxp, lg2, denp, NEDGE);
        // weighted message scatter
        hipLaunchKernelGGL(edge_agg_kernel, egrid, blk, 0, stream, ei_gp, lg0, mxp, denp, vtg,  aggp, NEDGE);
        hipLaunchKernelGGL(edge_agg_kernel, egrid, blk, 0, stream, ei_pg, lg1, mxg, deng, vtp1, aggg, NEDGE);
        hipLaunchKernelGGL(edge_agg_kernel, egrid, blk, 0, stream, ei_pp, lg2, mxp, denp, vtp2, aggp, NEDGE);

        // output conv: relu( sig(skip)*(gelu(agg)@Woc + b) + (1-sig)*x )
        hipLaunchKernelGGL((gemm_wmma<1,2>), ggrid(NGENE), blk, 0, stream,
            aggg, Woc_bf + (size_t)(l*2+0)*WSZ, b_oc + (size_t)(l*2+0)*HID, xgc, skip + l*2+0, xgn, NGENE, HID, HID);
        hipLaunchKernelGGL((gemm_wmma<1,2>), ggrid(NPEAK), blk, 0, stream,
            aggp, Woc_bf + (size_t)(l*2+1)*WSZ, b_oc + (size_t)(l*2+1)*HID, xpc, skip + l*2+1, xpn, NPEAK, HID, HID);

        cur = 1 - cur;
    }

    // ---- output projections into d_out (z_g then z_p) ----
    hipLaunchKernelGGL((gemm_wmma<0,0>), ggrid(NGENE), blk, 0, stream,
        xg[cur], Woutg_bf, b_out_g, nullptr, nullptr, out, NGENE, HID, HID);
    hipLaunchKernelGGL((gemm_wmma<0,0>), ggrid(NPEAK), blk, 0, stream,
        xp[cur], Woutp_bf, b_out_p, nullptr, nullptr, out + (size_t)NGENE * HID, NPEAK, HID, HID);
}
